// AdaptivePosePoolingv3_6339371729464
// MI455X (gfx1250) — matrix operand order repaired and checked by reading, not compiled, not run on hardware
//
#include <hip/hip_runtime.h>
#include <math.h>

// ---------------- problem constants ----------------
#define B_     2
#define T2_    9
#define T3_    27
#define P_     17
#define K_     9
#define C_     128
#define HEADS_ 8
#define DH_    16
#define T1_    (T2_*T3_)      // 243
#define PK_    (P_*K_)        // 153
#define N_     (B_*T2_)       // 18
#define LQ_    (T3_*PK_)      // 4131
#define ROWS_  (N_*LQ_)       // 74358  (= B*T1*PK)
#define ROWS0_ (B_*T1_*P_)    // 8262
#define LTOT_  9180
#define RVAL_  (N_*LTOT_)     // 165240

__device__ __constant__ int c_H[4] = {72, 36, 18, 9};
__device__ __constant__ int c_W[4] = {96, 48, 24, 12};
__device__ __constant__ int c_S[4] = {0, 6912, 8640, 9072};

// ---------------- WMMA types ----------------
typedef __attribute__((ext_vector_type(8)))  __bf16 v8bf;
typedef __attribute__((ext_vector_type(16))) __bf16 v16bf;
typedef __attribute__((ext_vector_type(8)))  float  v8f;

__device__ inline __bf16 f32_to_bf16(float f) {
  unsigned int u = __builtin_bit_cast(unsigned int, f);
  unsigned int r = (u + 0x7FFFu + ((u >> 16) & 1u)) >> 16;
  return __builtin_bit_cast(__bf16, (unsigned short)r);
}
__device__ inline float bf16u_to_f32(unsigned short u) {
  return __builtin_bit_cast(float, ((unsigned int)u) << 16);
}
__device__ inline float gelu_exact(float x) {
  return 0.5f * x * (1.0f + erff(x * 0.70710678118654752f));
}

#define WMMA_BF16(a, b, c) \
  __builtin_amdgcn_wmma_f32_16x16x32_bf16(false, (a), false, (b), (short)0, (c), false, false)

// A-fragment for 16x32 bf16: lane<16 -> K kk..kk+15; lane>=16 -> K kk+16..kk+31
__device__ inline v16bf load_a_frag(const __bf16* __restrict__ Ar, int half) {
  v8bf a0 = *(const v8bf*)(Ar + half * 16);
  v8bf a1 = *(const v8bf*)(Ar + 8 + half * 16);
  return __builtin_shufflevector(a0, a1, 0, 1, 2, 3, 4, 5, 6, 7,
                                 8, 9, 10, 11, 12, 13, 14, 15);
}

// =====================================================================
// Register-blocked GEMM on bf16-resident operands.
//   A  : bf16 [M, Kd] row-major ; WT : bf16 [Nc, Kd] (transposed weights)
//   One wave computes a 2x4 grid of 16x16 tiles (32 rows x 64 cols).
//   All 12 b128 fragment loads sit at the loop top (fresh variables, no
//   ping-pong copies); pointers bump by 32 elements so addresses fold to
//   immediate offsets under unrolling; the compiler pipelines across its
//   own unroll without accumulator spills.
//   out = act(A @ W + bias) (+ resid[f32]) -> f32 and/or bf16.
// Requires: Kd % 32 == 0, Nc % 64 == 0.
// =====================================================================
__global__ void gemm_bf16_wmma(const __bf16* __restrict__ A, const __bf16* __restrict__ WT,
                               const float* __restrict__ bias, const float* __restrict__ resid,
                               float* __restrict__ outf, __bf16* __restrict__ outb,
                               int M, int Kd, int Nc, int act) {
  const int lane = threadIdx.x;          // 0..31
  const int half = lane >> 4;            // 0 or 1
  const int l16  = lane & 15;
  const int rowBase = blockIdx.x * 32;
  const int colBase = blockIdx.y * 64;

  int r0 = rowBase + l16;      int rA0 = r0 < M ? r0 : (M - 1);
  int r1 = rowBase + 16 + l16; int rA1 = r1 < M ? r1 : (M - 1);
  const __bf16* pA0 = A + (size_t)rA0 * Kd;
  const __bf16* pA1 = A + (size_t)rA1 * Kd;
  // B 32x16: lane<16 -> K kk..kk+15 ; lane>=16 -> K kk+16..kk+31 (fold half*16 here)
  const __bf16* pW0 = WT + (size_t)(colBase + l16) * Kd + half * 16;
  const __bf16* pW1 = pW0 + (size_t)16 * Kd;
  const __bf16* pW2 = pW0 + (size_t)32 * Kd;
  const __bf16* pW3 = pW0 + (size_t)48 * Kd;

  v8f acc[2][4];
#pragma unroll
  for (int mt = 0; mt < 2; ++mt)
#pragma unroll
    for (int nt = 0; nt < 4; ++nt) acc[mt][nt] = (v8f){};

  for (int steps = Kd >> 5; steps > 0; --steps) {
    v16bf a0 = load_a_frag(pA0, half);
    v16bf a1 = load_a_frag(pA1, half);
    v16bf b0 = *(const v16bf*)(pW0);
    v16bf b1 = *(const v16bf*)(pW1);
    v16bf b2 = *(const v16bf*)(pW2);
    v16bf b3 = *(const v16bf*)(pW3);
    pA0 += 32; pA1 += 32; pW0 += 32; pW1 += 32; pW2 += 32; pW3 += 32;

    acc[0][0] = WMMA_BF16(a0, b0, acc[0][0]);
    acc[1][0] = WMMA_BF16(a1, b0, acc[1][0]);
    acc[0][1] = WMMA_BF16(a0, b1, acc[0][1]);
    acc[1][1] = WMMA_BF16(a1, b1, acc[1][1]);
    acc[0][2] = WMMA_BF16(a0, b2, acc[0][2]);
    acc[1][2] = WMMA_BF16(a1, b2, acc[1][2]);
    acc[0][3] = WMMA_BF16(a0, b3, acc[0][3]);
    acc[1][3] = WMMA_BF16(a1, b3, acc[1][3]);
  }

#pragma unroll
  for (int nt = 0; nt < 4; ++nt) {
    int col = colBase + nt * 16 + l16;
    float bv = bias ? bias[col] : 0.0f;
#pragma unroll
    for (int mt = 0; mt < 2; ++mt) {
#pragma unroll
      for (int r = 0; r < 8; ++r) {
        int orow = rowBase + mt * 16 + r + half * 8;
        if (orow < M) {
          float v = acc[mt][nt][r] + bv;
          if (act == 1) v = gelu_exact(v);
          if (resid) v += resid[(size_t)orow * Nc + col];
          if (outf) outf[(size_t)orow * Nc + col] = v;
          if (outb) outb[(size_t)orow * Nc + col] = f32_to_bf16(v);
        }
      }
    }
  }
}

// =====================================================================
// femb GEMM: fflat_bf[n, start+hw, d] = bf16(sum_c f[n,c,hw]*fw[c,d] + fb[d])
// A element (row=n*HW+hw, k=c) = f[(n*FCH+c)*HW + hw]  (channel-strided f32)
// =====================================================================
__global__ void femb_gemm_wmma(const float* __restrict__ f, const __bf16* __restrict__ fwT,
                               const float* __restrict__ fb, __bf16* __restrict__ fflat_bf,
                               int HW, int FCH, int start) {
  const int lane = threadIdx.x;
  const int half = lane >> 4;
  const int l16  = lane & 15;
  const int M = N_ * HW;

  int rowA = blockIdx.x * 16 + l16;
  int rA = rowA < M ? rowA : (M - 1);
  int nA = rA / HW, hwA = rA % HW;
  int colB = blockIdx.y * 16 + l16;
  const float* fr = f + (size_t)nA * FCH * HW + hwA;
  const __bf16* Wr = fwT + (size_t)colB * FCH + half * 16;

  v8f acc = {};
  for (int kk = 0; kk < FCH; kk += 32) {
    v16bf a, b;
#pragma unroll
    for (int i = 0; i < 16; ++i) {
      int k = kk + i + half * 16;
      a[i] = f32_to_bf16(fr[(size_t)k * HW]);
    }
    b = *(const v16bf*)(Wr + kk);
    acc = WMMA_BF16(a, b, acc);
  }
  float bv = fb[colB];
#pragma unroll
  for (int r = 0; r < 8; ++r) {
    int orow = blockIdx.x * 16 + r + half * 8;
    if (orow < M) {
      int n = orow / HW, hw = orow % HW;
      fflat_bf[((size_t)n * LTOT_ + start + hw) * C_ + colB] = f32_to_bf16(acc[r] + bv);
    }
  }
}

// transpose + convert weights: WT[n*Kd + k] = bf16(W[k*Nc + n])
__global__ void packT_k(const float* __restrict__ W, __bf16* __restrict__ WT,
                        int Kd, int Ncol) {
  int i = blockIdx.x * blockDim.x + threadIdx.x;
  if (i >= Kd * Ncol) return;
  int n = i / Kd, k = i % Kd;
  WT[i] = f32_to_bf16(W[(size_t)k * Ncol + n]);
}

// pack kr_w (C,C,K) transposed: WT[o, j=k*C+c] = kr_w[o, c, k]
__global__ void pack_krT_k(const float* __restrict__ kr_w, __bf16* __restrict__ WT) {
  int i = blockIdx.x * blockDim.x + threadIdx.x;
  if (i >= C_ * K_ * C_) return;
  int o = i / (K_ * C_), j = i % (K_ * C_);
  int k = j / C_, c = j % C_;
  WT[i] = f32_to_bf16(kr_w[((size_t)o * C_ + c) * K_ + k]);
}

__global__ void convert_bf_k(const float* __restrict__ in, __bf16* __restrict__ out, int total) {
  int i = blockIdx.x * blockDim.x + threadIdx.x;
  if (i < total) out[i] = f32_to_bf16(in[i]);
}

// =====================================================================
// Bilinear sample from bf16 channel-last feature block, zero padding
// =====================================================================
__device__ inline float bilin_bf(const unsigned short* __restrict__ base, int Hh, int Ww,
                                 float sx, float sy, int c) {
  float x0f = floorf(sx), y0f = floorf(sy);
  int x0 = (int)x0f, y0 = (int)y0f;
  float wx1 = sx - x0f, wy1 = sy - y0f;
  float r = 0.0f;
#pragma unroll
  for (int dy = 0; dy < 2; ++dy) {
#pragma unroll
    for (int dx = 0; dx < 2; ++dx) {
      int xx = x0 + dx, yy = y0 + dy;
      if (xx >= 0 && xx < Ww && yy >= 0 && yy < Hh) {
        float wgt = (dy ? wy1 : 1.0f - wy1) * (dx ? wx1 : 1.0f - wx1);
        r += wgt * bf16u_to_f32(base[((size_t)yy * Ww + xx) * C_ + c]);
      }
    }
  }
  return r;
}

__device__ inline void decode_ref(const float* __restrict__ ref, int n, int q,
                                  float& rx, float& ry) {
  int b = n / T2_, t2v = n % T2_;
  int t3 = q / PK_, pk = q % PK_;
  int p = pk / K_, kk = pk % K_;
  int t1 = t2v * T3_ + t3;
  const float* rp = ref + (((size_t)(b * T1_ + t1)) * P_ + p) * 2;
  rx = rp[0] + (float)(kk % 3 - 1);
  ry = rp[1] + (float)(kk / 3 - 1);
}

// =====================================================================
// init feature sampling: x = mean_l gridsample(femb_l, init_grid) + pe_x
// =====================================================================
__global__ void init_sample_k(const unsigned short* __restrict__ fflat_bf,
                              const float* __restrict__ ref,
                              const float* __restrict__ pe_x, float* __restrict__ x) {
  int row = blockIdx.x;                 // n*LQ_ + q
  int c = threadIdx.x;                  // 0..127
  int n = row / LQ_, q = row % LQ_;
  float rx, ry;
  decode_ref(ref, n, q, rx, ry);
  float acc = 0.0f;
#pragma unroll
  for (int l = 0; l < 4; ++l) {
    int Hl = c_H[l], Wl = c_W[l];
    const unsigned short* base = fflat_bf + ((size_t)n * LTOT_ + c_S[l]) * C_;
    float sx = rx * ((float)Wl / 288.0f) - 0.5f;
    float sy = ry * ((float)Hl / 384.0f) - 0.5f;
    acc += bilin_bf(base, Hl, Wl, sx, sy, c);
  }
  acc *= 0.25f;
  int t2v = n % T2_;
  acc += pe_x[((size_t)t2v * LQ_ + q) * C_ + c];
  x[(size_t)row * C_ + c] = acc;
}

// =====================================================================
// layernorm over C=128; writes bf16 and/or f32
// =====================================================================
__global__ void layernorm_k(const float* __restrict__ x, const float* __restrict__ w,
                            const float* __restrict__ b, __bf16* __restrict__ outb,
                            float* __restrict__ outf) {
  __shared__ float sh[128];
  int row = blockIdx.x, t = threadIdx.x;
  float v = x[(size_t)row * C_ + t];
  sh[t] = v; __syncthreads();
  for (int s = 64; s > 0; s >>= 1) { if (t < s) sh[t] += sh[t + s]; __syncthreads(); }
  float mean = sh[0] * (1.0f / 128.0f); __syncthreads();
  float dv = v - mean;
  sh[t] = dv * dv; __syncthreads();
  for (int s = 64; s > 0; s >>= 1) { if (t < s) sh[t] += sh[t + s]; __syncthreads(); }
  float var = sh[0] * (1.0f / 128.0f);
  float o = dv * rsqrtf(var + 1e-5f) * w[t] + b[t];
  if (outb) outb[(size_t)row * C_ + t] = f32_to_bf16(o);
  if (outf) outf[(size_t)row * C_ + t] = o;
}

// per-head softmax over groups of 16 within a row of 128 (in-place f32)
__global__ void softmax_head_k(float* __restrict__ aw) {
  __shared__ float sh[128];
  int row = blockIdx.x, t = threadIdx.x;
  float v = aw[(size_t)row * C_ + t];
  sh[t] = v; __syncthreads();
  int base = t & ~15;
  float mx = -3.4e38f;
#pragma unroll
  for (int j = 0; j < 16; ++j) mx = fmaxf(mx, sh[base + j]);
  float e = expf(v - mx);
  __syncthreads();
  sh[t] = e; __syncthreads();
  float sum = 0.0f;
#pragma unroll
  for (int j = 0; j < 16; ++j) sum += sh[base + j];
  aw[(size_t)row * C_ + t] = e / sum;
}

// =====================================================================
// deformable sampling (bf16 gathers) + weighted accumulation -> bf16
// =====================================================================
__global__ void msdeform_sample_k(const unsigned short* __restrict__ val_bf,
                                  const float* __restrict__ off,
                                  const float* __restrict__ aw, const float* __restrict__ ref,
                                  __bf16* __restrict__ out) {
  int row = blockIdx.x;
  int t = threadIdx.x;                  // channel = t, h = t/16
  int n = row / LQ_, q = row % LQ_;
  float rx, ry;
  decode_ref(ref, n, q, rx, ry);
  float dref_x = rx * (1.0f / 288.0f);
  float dref_y = ry * (1.0f / 384.0f);
  int h = t >> 4;
  float acc = 0.0f;
#pragma unroll
  for (int l = 0; l < 4; ++l) {
    int Hl = c_H[l], Wl = c_W[l];
    const unsigned short* base = val_bf + ((size_t)n * LTOT_ + c_S[l]) * C_;
#pragma unroll
    for (int pt = 0; pt < 4; ++pt) {
      int oi = (h * 4 + l) * 4 + pt;
      float ox = off[(size_t)row * 256 + oi * 2 + 0];
      float oy = off[(size_t)row * 256 + oi * 2 + 1];
      float a  = aw[(size_t)row * 128 + oi];
      float sx = dref_x * (float)Wl + ox - 0.5f;
      float sy = dref_y * (float)Hl + oy - 0.5f;
      acc += a * bilin_bf(base, Hl, Wl, sx, sy, t);
    }
  }
  out[(size_t)row * C_ + t] = f32_to_bf16(acc);
}

// x0 = x_0 + pe_x0 -> bf16 (only consumed by K/V GEMMs)
__global__ void add_pe_k(const float* __restrict__ a, const float* __restrict__ pe,
                         __bf16* __restrict__ out, int total, int period) {
  int i = blockIdx.x * blockDim.x + threadIdx.x;
  if (i < total) out[i] = f32_to_bf16(a[i] + pe[i % period]);
}

// =====================================================================
// MHA attention: per (b*T1, query): 8 heads x 17 keys, DH=16 -> bf16 out
// =====================================================================
__global__ void mha_att_k(const float* __restrict__ Q, const float* __restrict__ Kb,
                          const float* __restrict__ Vb, __bf16* __restrict__ out) {
  __shared__ float qs[128];
  __shared__ float sc[HEADS_][P_];
  int row = blockIdx.x;
  int np = row / PK_;
  int t = threadIdx.x;
  qs[t] = Q[(size_t)row * C_ + t];
  __syncthreads();
  for (int s = t; s < HEADS_ * P_; s += 128) {
    int h = s / P_, kk = s % P_;
    const float* kr = Kb + ((size_t)np * P_ + kk) * C_ + h * DH_;
    float dot = 0.0f;
#pragma unroll
    for (int d = 0; d < DH_; ++d) dot += qs[h * DH_ + d] * kr[d];
    sc[h][kk] = dot * 0.25f;            // 1/sqrt(16)
  }
  __syncthreads();
  int h = t >> 4, d = t & 15;
  float mx = -3.4e38f;
#pragma unroll
  for (int kk = 0; kk < P_; ++kk) mx = fmaxf(mx, sc[h][kk]);
  float sum = 0.0f;
#pragma unroll
  for (int kk = 0; kk < P_; ++kk) sum += expf(sc[h][kk] - mx);
  float inv = 1.0f / sum;
  float o = 0.0f;
#pragma unroll
  for (int kk = 0; kk < P_; ++kk) {
    float a = expf(sc[h][kk] - mx) * inv;
    o += a * Vb[((size_t)np * P_ + kk) * C_ + h * DH_ + d];
  }
  out[(size_t)row * C_ + t] = f32_to_bf16(o);
}

// head: out = tanh(h @ h1_w + h1_b) @ h2_w + h2_b
__global__ void head_k(const float* __restrict__ h, const float* __restrict__ h1w,
                       const float* __restrict__ h1b, const float* __restrict__ h2w,
                       const float* __restrict__ h2b, float* __restrict__ out) {
  __shared__ float hid[64];
  int row = blockIdx.x, t = threadIdx.x;
  float acc = h1b[t];
  for (int c = 0; c < C_; ++c) acc += h[(size_t)row * C_ + c] * h1w[c * 64 + t];
  hid[t] = tanhf(acc);
  __syncthreads();
  if (t < 3) {
    float o = h2b[t];
#pragma unroll
    for (int j = 0; j < 64; ++j) o += hid[j] * h2w[j * 3 + t];
    out[(size_t)row * 3 + t] = o;
  }
}

// =====================================================================
// host-side launcher
// =====================================================================
static inline void launch_gemm(const __bf16* A, const __bf16* WT, const float* bias,
                               const float* resid, float* outf, __bf16* outb,
                               int M, int Kd, int Ncol, int act, hipStream_t s) {
  dim3 g((M + 31) / 32, Ncol / 64);
  gemm_bf16_wmma<<<g, dim3(32), 0, s>>>(A, WT, bias, resid, outf, outb, M, Kd, Ncol, act);
}

static inline void launch_packT(const float* W, __bf16* WT, int Kd, int Ncol, hipStream_t s) {
  int total = Kd * Ncol;
  packT_k<<<(total + 255) / 256, 256, 0, s>>>(W, WT, Kd, Ncol);
}

extern "C" void kernel_launch(void* const* d_in, const int* in_sizes, int n_in,
                              void* d_out, int out_size, void* d_ws, size_t ws_size,
                              hipStream_t stream) {
  const float* fin[4] = {(const float*)d_in[0], (const float*)d_in[1],
                         (const float*)d_in[2], (const float*)d_in[3]};
  const float* x_0   = (const float*)d_in[4];
  const float* ref   = (const float*)d_in[5];
  const float* pe_x  = (const float*)d_in[6];
  const float* pe_x0 = (const float*)d_in[7];
  const float* fw[4] = {(const float*)d_in[8],  (const float*)d_in[10],
                        (const float*)d_in[12], (const float*)d_in[14]};
  const float* fb[4] = {(const float*)d_in[9],  (const float*)d_in[11],
                        (const float*)d_in[13], (const float*)d_in[15]};
  const float* so_w = (const float*)d_in[16]; const float* so_b = (const float*)d_in[17];
  const float* aw_w = (const float*)d_in[18]; const float* aw_b = (const float*)d_in[19];
  const float* vp_w = (const float*)d_in[20]; const float* vp_b = (const float*)d_in[21];
  const float* op_w = (const float*)d_in[22]; const float* op_b = (const float*)d_in[23];
  const float* qkv_w = (const float*)d_in[24]; const float* qkv_b = (const float*)d_in[25];
  const float* co_w = (const float*)d_in[26]; const float* co_b = (const float*)d_in[27];
  const float* n1_w = (const float*)d_in[28]; const float* n1_b = (const float*)d_in[29];
  const float* n2_w = (const float*)d_in[30]; const float* n2_b = (const float*)d_in[31];
  const float* n3_w = (const float*)d_in[32]; const float* n3_b = (const float*)d_in[33];
  const float* m1_w = (const float*)d_in[34]; const float* m1_b = (const float*)d_in[35];
  const float* m2_w = (const float*)d_in[36]; const float* m2_b = (const float*)d_in[37];
  const float* kr_w = (const float*)d_in[38]; const float* kr_b = (const float*)d_in[39];
  const float* hn_w = (const float*)d_in[40]; const float* hn_b = (const float*)d_in[41];
  const float* h1_w = (const float*)d_in[42]; const float* h1_b = (const float*)d_in[43];
  const float* h2_w = (const float*)d_in[44]; const float* h2_b = (const float*)d_in[45];
  float* outp = (float*)d_out;

  // ---------------- workspace layout (float units, all 256B aligned) ----
  float* ws = (float*)d_ws;
  float* x     = ws;                                      // ROWS_*128 f32
  float* offb  = x     + (size_t)ROWS_ * C_;              // ROWS_*256 f32 (aliased by Q f32)
  float* awb   = offb  + (size_t)ROWS_ * 256;             // ROWS_*128 f32
  float* kbuf  = awb   + (size_t)ROWS_ * C_;              // ROWS0_*128 f32
  float* vbuf  = kbuf  + (size_t)ROWS0_ * C_;             // ROWS0_*128 f32
  float* xr    = vbuf  + (size_t)ROWS0_ * C_;             // ROWS0_*128 f32
  float* shreg = xr    + (size_t)ROWS0_ * C_;             // ROWS_*512 bf16 region
  __bf16* val_bf   = (__bf16*)shreg;                      //  alias 1: RVAL_*128 bf16
  __bf16* hid_bf   = (__bf16*)shreg;                      //  alias 2: ROWS_*512 bf16
  float* nxt   = shreg + (size_t)ROWS_ * 256;             // (ROWS_*512 bf16 = ROWS_*256 f32)
  __bf16* xn_bf    = (__bf16*)nxt;                        // ROWS_*128 bf16
  __bf16* fflat_bf = xn_bf + (size_t)ROWS_ * C_;          // RVAL_*128 bf16
  __bf16* tb_bf    = fflat_bf + (size_t)RVAL_ * C_;       // ROWS_*128 bf16 (sampler out / x_bf)
  __bf16* x0_bf    = tb_bf + (size_t)ROWS_ * C_;          // ROWS0_*128 bf16
  __bf16* wkrT     = x0_bf + (size_t)ROWS0_ * C_;         // 128*1152 bf16
  __bf16* wT       = wkrT + (size_t)C_ * K_ * C_;         // packed weights ~1.18M bf16
  float*  qf32     = offb;                                // Q aliases off region

  const int FCHs[4] = {32, 64, 128, 256};
  const int HWs[4] = {72 * 96, 36 * 48, 18 * 24, 9 * 12};
  size_t fwT_off[4]; size_t acc_off = 0;
  for (int l = 0; l < 4; ++l) { fwT_off[l] = acc_off; acc_off += (size_t)FCHs[l] * C_; }
  const size_t perLayer = 32768 + 16384 * 6 + 65536 * 2;
  // layer block: so:0 aw:32768 vp:49152 op:65536 q:81920 k:98304 v:114688 co:131072 m1:147456 m2:212992

  // ---- weight packing ----
  for (int l = 0; l < 4; ++l) launch_packT(fw[l], wT + fwT_off[l], FCHs[l], C_, stream);
  for (int i = 0; i < 4; ++i) {
    __bf16* base = wT + acc_off + (size_t)i * perLayer;
    launch_packT(so_w + (size_t)i * C_ * 256, base + 0,      C_, 256, stream);
    launch_packT(aw_w + (size_t)i * C_ * C_,  base + 32768,  C_, C_, stream);
    launch_packT(vp_w + (size_t)i * C_ * C_,  base + 49152,  C_, C_, stream);
    launch_packT(op_w + (size_t)i * C_ * C_,  base + 65536,  C_, C_, stream);
    launch_packT(qkv_w + (size_t)(i * 3 + 0) * C_ * C_, base + 81920,  C_, C_, stream);
    launch_packT(qkv_w + (size_t)(i * 3 + 1) * C_ * C_, base + 98304,  C_, C_, stream);
    launch_packT(qkv_w + (size_t)(i * 3 + 2) * C_ * C_, base + 114688, C_, C_, stream);
    launch_packT(co_w + (size_t)i * C_ * C_,  base + 131072, C_, C_, stream);
    launch_packT(m1_w + (size_t)i * C_ * 512, base + 147456, C_, 512, stream);
    launch_packT(m2_w + (size_t)i * 512 * C_, base + 212992, 512, C_, stream);
  }
  {
    int total = C_ * K_ * C_;
    pack_krT_k<<<(total + 255) / 256, 256, 0, stream>>>(kr_w, wkrT);
  }

  // 1) feature embedding GEMMs -> fflat_bf
  {
    const int STARTs[4] = {0, 6912, 8640, 9072};
    for (int l = 0; l < 4; ++l) {
      int M = N_ * HWs[l];
      dim3 g((M + 15) / 16, C_ / 16);
      femb_gemm_wmma<<<g, dim3(32), 0, stream>>>(fin[l], wT + fwT_off[l], fb[l],
                                                 fflat_bf, HWs[l], FCHs[l], STARTs[l]);
    }
  }

  // 2) initial multi-level sampling + mean + pe_x -> x
  init_sample_k<<<ROWS_, 128, 0, stream>>>((const unsigned short*)fflat_bf, ref, pe_x, x);

  // 3) x0 = x_0 + pe_x0 -> bf16
  {
    int total = ROWS0_ * C_, period = T1_ * P_ * C_;
    add_pe_k<<<(total + 255) / 256, 256, 0, stream>>>(x_0, pe_x0, x0_bf, total, period);
  }

  // 4) decoder layers
  for (int i = 0; i < 4; ++i) {
    __bf16* Wb = wT + acc_off + (size_t)i * perLayer;

    // ---- msdeform block ----
    layernorm_k<<<ROWS_, 128, 0, stream>>>(x, n1_w + i * C_, n1_b + i * C_, xn_bf, nullptr);
    launch_gemm(xn_bf, Wb + 0,     so_b + (size_t)i * 256, nullptr, offb, nullptr,
                ROWS_, C_, 256, 0, stream);
    launch_gemm(xn_bf, Wb + 32768, aw_b + (size_t)i * C_, nullptr, awb, nullptr,
                ROWS_, C_, C_, 0, stream);
    softmax_head_k<<<ROWS_, 128, 0, stream>>>(awb);
    launch_gemm(fflat_bf, Wb + 49152, vp_b + (size_t)i * C_, nullptr, nullptr, val_bf,
                RVAL_, C_, C_, 0, stream);
    msdeform_sample_k<<<ROWS_, 128, 0, stream>>>((const unsigned short*)val_bf, offb, awb,
                                                 ref, tb_bf);
    launch_gemm(tb_bf, Wb + 65536, op_b + (size_t)i * C_, x, x, nullptr,
                ROWS_, C_, C_, 0, stream);

    // ---- cross MHA block ----
    layernorm_k<<<ROWS_, 128, 0, stream>>>(x, n2_w + i * C_, n2_b + i * C_, xn_bf, nullptr);
    launch_gemm(xn_bf, Wb + 81920,  qkv_b + (size_t)(i * 3 + 0) * C_, nullptr, qf32, nullptr,
                ROWS_, C_, C_, 0, stream);
    launch_gemm(x0_bf, Wb + 98304,  qkv_b + (size_t)(i * 3 + 1) * C_, nullptr, kbuf, nullptr,
                ROWS0_, C_, C_, 0, stream);
    launch_gemm(x0_bf, Wb + 114688, qkv_b + (size_t)(i * 3 + 2) * C_, nullptr, vbuf, nullptr,
                ROWS0_, C_, C_, 0, stream);
    mha_att_k<<<ROWS_, 128, 0, stream>>>(qf32, kbuf, vbuf, xn_bf);
    launch_gemm(xn_bf, Wb + 131072, co_b + (size_t)i * C_, x, x, nullptr,
                ROWS_, C_, C_, 0, stream);

    // ---- MLP block ----
    layernorm_k<<<ROWS_, 128, 0, stream>>>(x, n3_w + i * C_, n3_b + i * C_, xn_bf, nullptr);
    launch_gemm(xn_bf, Wb + 147456, m1_b + (size_t)i * 512, nullptr, nullptr, hid_bf,
                ROWS_, C_, 512, 1 /*gelu*/, stream);
    launch_gemm(hid_bf, Wb + 212992, m2_b + (size_t)i * C_, x, x, nullptr,
                ROWS_, 512, C_, 0, stream);
  }

  // 5) head
  {
    int total = ROWS_ * C_;
    convert_bf_k<<<(total + 255) / 256, 256, 0, stream>>>(x, tb_bf, total);
  }
  launch_gemm(tb_bf, wkrT, kr_b, nullptr, xr, nullptr, ROWS0_, K_ * C_, C_, 0, stream);
  layernorm_k<<<ROWS0_, 128, 0, stream>>>(xr, hn_w, hn_b, nullptr, kbuf);
  head_k<<<ROWS0_, 64, 0, stream>>>(kbuf, h1_w, h1_b, h2_w, h2_b, outp);
}
